// GriffinBlock1D_53987738910746
// MI455X (gfx1250) — compile-verified, matrix operand order checked by reading, not run on hardware
//
#include <hip/hip_runtime.h>
#include <hip/hip_bf16.h>
#include <math.h>

// ---------------------------------------------------------------------------
// Griffin block for MI455X (gfx1250, wave32, WMMA).
// All dense GEMMs use v_wmma_f32_16x16x32_bf16 (bf16 A/B, f32 accumulate).
// One wave computes a 32(M) x 64(N) tile: 2 A frags x 4 B frags = 8 WMMAs
// per 32-deep K-step; B is reused across both M-tiles, A across all N-tiles
// (~22 FLOP/byte of cache traffic, load:wmma ratio 1.5).
// Scan / LN / windowed attention run in f32 VALU (negligible FLOP share).
// ---------------------------------------------------------------------------

typedef unsigned short u16;
typedef __attribute__((ext_vector_type(16))) __bf16    v16bf;
typedef __attribute__((ext_vector_type(8)))  float     v8f;
typedef __attribute__((ext_vector_type(4)))  unsigned  v4u;

union FragU { v4u q[2]; v16bf v; };

__device__ __forceinline__ u16 f2b(float f) {
    unsigned u = __builtin_bit_cast(unsigned, f);
    u = (u + 0x7FFFu + ((u >> 16) & 1u)) >> 16;        // round-to-nearest-even
    return (u16)u;
}

// --------------------------- f32 -> bf16 convert ---------------------------
__global__ __launch_bounds__(256) void cvt_bf16_kernel(
    const float* __restrict__ in, u16* __restrict__ out, int n)
{
    int i = blockIdx.x * 256 + threadIdx.x;
    if (i < n) out[i] = f2b(in[i]);
}

// ------------------------------- WMMA GEMM ---------------------------------
// out[m][n] = sum_k A[m][k] * W[n][k]   (A: MxK bf16, W: NxK bf16, both row-major)
// modes: 0 plain f32*scale | 1 split sigmoid/bf16 (W_in) | 2 +res f32
//        3 +bias, GELU, bf16 | 4 +res +bias f32
__global__ __launch_bounds__(256) void gemm_bf16_wmma(
    const u16* __restrict__ A, const u16* __restrict__ W,
    int M, int N, int K, int mode, float scale,
    const float* __restrict__ res, const float* __restrict__ bias,
    float* __restrict__ outf, u16* __restrict__ outb)
{
    const int lane   = threadIdx.x & 31;
    const int tilesN = N >> 6;                           // 64-wide strips
    const int wid    = blockIdx.x * 8 + (threadIdx.x >> 5);
    if (wid >= (M >> 5) * tilesN) return;                // 32-row strips
    const int m0   = (wid / tilesN) << 5;
    const int n0   = (wid % tilesN) << 6;
    const int half = lane >> 4;                          // K-half select (ISA A layout)
    const int lr   = lane & 15;                          // row (A) / col (B,D)

    v8f acc[2][4];
#pragma unroll
    for (int i = 0; i < 2; ++i)
#pragma unroll
        for (int j = 0; j < 4; ++j)
#pragma unroll
            for (int e = 0; e < 8; ++e) acc[i][j][e] = 0.0f;

    const u16* aRow0 = A + (size_t)(m0 + lr) * K + 8 * half;
    const u16* aRow1 = A + (size_t)(m0 + 16 + lr) * K + 8 * half;
#pragma unroll 1
    for (int k0 = 0; k0 < K; k0 += 32) {
        FragU a0, a1;
        a0.q[0] = *(const v4u*)(aRow0 + k0);             // K = k0 + 8*half + 0..7
        a0.q[1] = *(const v4u*)(aRow0 + k0 + 16);        // K = k0 + 16 + 8*half + 0..7
        a1.q[0] = *(const v4u*)(aRow1 + k0);
        a1.q[1] = *(const v4u*)(aRow1 + k0 + 16);
#pragma unroll
        for (int j = 0; j < 4; ++j) {
            const u16* bRow = W + (size_t)(n0 + 16 * j + lr) * K + 8 * half + k0;
            FragU b;
            b.q[0] = *(const v4u*)(bRow);
            b.q[1] = *(const v4u*)(bRow + 16);
            acc[0][j] = __builtin_amdgcn_wmma_f32_16x16x32_bf16(
                false, a0.v, false, b.v, (short)0, acc[0][j], false, false);
            acc[1][j] = __builtin_amdgcn_wmma_f32_16x16x32_bf16(
                false, a1.v, false, b.v, (short)0, acc[1][j], false, false);
        }
    }

#pragma unroll
    for (int i = 0; i < 2; ++i) {
#pragma unroll
        for (int j = 0; j < 4; ++j) {
            const int n = n0 + 16 * j + lr;
#pragma unroll
            for (int r = 0; r < 8; ++r) {
                const int row = m0 + 16 * i + 8 * half + r;  // D layout: m = r + 8*half
                float val = acc[i][j][r] * scale;
                if (mode == 0) {
                    outf[(size_t)row * N + n] = val;
                } else if (mode == 1) {                   // W_in: u=sigmoid | g2=bf16
                    if (n < 512) outf[(size_t)row * 512 + n] = 1.0f / (1.0f + __expf(-val));
                    else         outb[(size_t)row * 512 + (n - 512)] = f2b(val);
                } else if (mode == 2) {
                    outf[(size_t)row * N + n] = val + res[(size_t)row * N + n];
                } else if (mode == 3) {
                    float t = val + bias[n];
                    outb[(size_t)row * N + n] =
                        f2b(0.5f * t * (1.0f + erff(t * 0.70710678118654752f)));
                } else {                                  // mode 4
                    outf[(size_t)row * N + n] = val + res[(size_t)row * N + n] + bias[n];
                }
            }
        }
    }
}

// ---------------------------- GLRU sequential scan --------------------------
// h_t = u_t*h + (1-u_t)*c_t, per (b,d) channel; C is read+overwritten with y.
__global__ __launch_bounds__(256) void glru_scan_kernel(
    const float* __restrict__ U, float* C)
{
    const int idx = blockIdx.x * 256 + threadIdx.x;      // 0..2047 = b*512+d
    const int b = idx >> 9, d = idx & 511;
    size_t base = ((size_t)b * 1024) * 512 + d;
    float h = 0.0f;
    for (int t = 0; t < 1024; ++t) {
        size_t o = base + (size_t)t * 512;
        float u = U[o], c = C[o];
        h = u * h + (1.0f - u) * c;
        C[o] = h;                                        // y
    }
}

// ------------------------------- LayerNorm ----------------------------------
// One wave per token (wave32, 16 elems/lane, shfl_xor tree reduce).
// Optionally applies a second LN (g2/b2) before the bf16 store (fuses
// x2 = LN(pre) and ffn_in = LN(x2) into one kernel).
__global__ __launch_bounds__(256) void ln_kernel(
    const float* __restrict__ X, const float* __restrict__ g, const float* __restrict__ bta,
    float* __restrict__ outf, u16* __restrict__ outb,
    const float* __restrict__ g2, const float* __restrict__ b2)
{
    const int lane = threadIdx.x & 31;
    const int tok  = blockIdx.x * 8 + (threadIdx.x >> 5);
    const int gi   = lane * 16;
    const float* xp = X + (size_t)tok * 512 + gi;

    float xv[16];
#pragma unroll
    for (int i = 0; i < 16; ++i) xv[i] = xp[i];

    float s = 0.0f;
#pragma unroll
    for (int i = 0; i < 16; ++i) s += xv[i];
#pragma unroll
    for (int off = 16; off; off >>= 1) s += __shfl_xor(s, off, 32);
    float mean = s * (1.0f / 512.0f);

    float vs = 0.0f;
#pragma unroll
    for (int i = 0; i < 16; ++i) { float d = xv[i] - mean; vs += d * d; }
#pragma unroll
    for (int off = 16; off; off >>= 1) vs += __shfl_xor(vs, off, 32);
    float rstd = rsqrtf(vs * (1.0f / 512.0f) + 1e-5f);

    float yv[16];
#pragma unroll
    for (int i = 0; i < 16; ++i)
        yv[i] = (xv[i] - mean) * rstd * g[gi + i] + bta[gi + i];

    if (outf) {
#pragma unroll
        for (int i = 0; i < 16; ++i) outf[(size_t)tok * 512 + gi + i] = yv[i];
    }
    if (outb) {
        if (g2) {                                         // fused second LN
            float s2 = 0.0f;
#pragma unroll
            for (int i = 0; i < 16; ++i) s2 += yv[i];
#pragma unroll
            for (int off = 16; off; off >>= 1) s2 += __shfl_xor(s2, off, 32);
            float m2 = s2 * (1.0f / 512.0f);
            float v2 = 0.0f;
#pragma unroll
            for (int i = 0; i < 16; ++i) { float d = yv[i] - m2; v2 += d * d; }
#pragma unroll
            for (int off = 16; off; off >>= 1) v2 += __shfl_xor(v2, off, 32);
            float r2 = rsqrtf(v2 * (1.0f / 512.0f) + 1e-5f);
#pragma unroll
            for (int i = 0; i < 16; ++i)
                yv[i] = (yv[i] - m2) * r2 * g2[gi + i] + b2[gi + i];
        }
#pragma unroll
        for (int i = 0; i < 16; ++i) outb[(size_t)tok * 512 + gi + i] = f2b(yv[i]);
    }
}

// ------------------------- windowed attention (+-16) ------------------------
// One wave per (b,t,h); head dim 128 = 4 f32/lane; 33 scores live in registers.
// q is pre-scaled by d^-0.5 in its GEMM epilogue.
__global__ __launch_bounds__(256) void local_attn_kernel(
    const float* __restrict__ Q, const float* __restrict__ Kt,
    const float* __restrict__ V, u16* __restrict__ out)
{
    const int lane = threadIdx.x & 31;
    const int wid  = blockIdx.x * 8 + (threadIdx.x >> 5); // 0..16383
    const int h = wid & 3, t = (wid >> 2) & 1023, b = wid >> 12;
    const size_t rowbase = ((size_t)(b * 1024 + t)) * 512 + h * 128 + lane;

    float qv[4];
#pragma unroll
    for (int i = 0; i < 4; ++i) qv[i] = Q[rowbase + 32 * i];

    float sc[33];
#pragma unroll
    for (int w = 0; w < 33; ++w) {
        const int j = t + w - 16;                         // wave-uniform validity
        float s = -3.0e38f;
        if ((unsigned)j < 1024u) {
            const float* kp = Kt + ((size_t)(b * 1024 + j)) * 512 + h * 128 + lane;
            s = qv[0] * kp[0] + qv[1] * kp[32] + qv[2] * kp[64] + qv[3] * kp[96];
#pragma unroll
            for (int off = 16; off; off >>= 1) s += __shfl_xor(s, off, 32);
        }
        sc[w] = s;
    }

    float mx = sc[0];
#pragma unroll
    for (int w = 1; w < 33; ++w) mx = fmaxf(mx, sc[w]);
    float ssum = 0.0f;
#pragma unroll
    for (int w = 0; w < 33; ++w) { sc[w] = __expf(sc[w] - mx); ssum += sc[w]; }
    const float inv = 1.0f / ssum;

    float acc[4] = {0.0f, 0.0f, 0.0f, 0.0f};
#pragma unroll
    for (int w = 0; w < 33; ++w) {
        const int j = t + w - 16;
        if ((unsigned)j < 1024u) {
            const float* vp = V + ((size_t)(b * 1024 + j)) * 512 + h * 128 + lane;
            const float p = sc[w];
#pragma unroll
            for (int i = 0; i < 4; ++i) acc[i] += p * vp[32 * i];
        }
    }
#pragma unroll
    for (int i = 0; i < 4; ++i) out[rowbase + 32 * i] = f2b(acc[i] * inv);
}

// ------------------------------- launcher -----------------------------------
extern "C" void kernel_launch(void* const* d_in, const int* in_sizes, int n_in,
                              void* d_out, int out_size, void* d_ws, size_t ws_size,
                              hipStream_t stream)
{
    (void)in_sizes; (void)n_in; (void)out_size; (void)ws_size;
    const float* x       = (const float*)d_in[0];
    const float* W_in    = (const float*)d_in[1];
    const float* W_state = (const float*)d_in[2];
    const float* glru_g  = (const float*)d_in[3];
    const float* glru_b  = (const float*)d_in[4];
    const float* Wq      = (const float*)d_in[5];
    const float* Wk      = (const float*)d_in[6];
    const float* Wv      = (const float*)d_in[7];
    const float* Wp      = (const float*)d_in[8];
    const float* attn_g  = (const float*)d_in[9];
    const float* attn_b  = (const float*)d_in[10];
    const float* ffn_g   = (const float*)d_in[11];
    const float* ffn_b   = (const float*)d_in[12];
    const float* W1      = (const float*)d_in[13];
    const float* b1      = (const float*)d_in[14];
    const float* W2      = (const float*)d_in[15];
    const float* b2      = (const float*)d_in[16];
    const float* out_g   = (const float*)d_in[17];
    const float* out_b   = (const float*)d_in[18];
    float* out = (float*)d_out;

    const int M = 4096;                       // B*T
    size_t off = 0;
    auto alloc = [&](size_t bytes) -> void* {
        void* p = (char*)d_ws + off;
        off += (bytes + 255) & ~(size_t)255;
        return p;
    };
    u16*   xb   = (u16*)  alloc((size_t)M * 512 * 2);
    u16*   Winb = (u16*)  alloc((size_t)1024 * 512 * 2);
    u16*   Wstb = (u16*)  alloc((size_t)512 * 512 * 2);
    u16*   Wqb  = (u16*)  alloc((size_t)512 * 512 * 2);
    u16*   Wkb  = (u16*)  alloc((size_t)512 * 512 * 2);
    u16*   Wvb  = (u16*)  alloc((size_t)512 * 512 * 2);
    u16*   Wpb  = (u16*)  alloc((size_t)512 * 512 * 2);
    u16*   W1b  = (u16*)  alloc((size_t)2048 * 512 * 2);
    u16*   W2b  = (u16*)  alloc((size_t)512 * 2048 * 2);
    float* ubuf = (float*)alloc((size_t)M * 512 * 4);    // sigmoid gate
    u16*   g2b  = (u16*)  alloc((size_t)M * 512 * 2);    // gv[:, D:] bf16
    float* cand = (float*)alloc((size_t)M * 512 * 4);    // cand, then y (in place)
    float* x1   = (float*)alloc((size_t)M * 512 * 4);
    u16*   x1b  = (u16*)  alloc((size_t)M * 512 * 2);
    float* q    = (float*)alloc((size_t)M * 512 * 4);
    float* k    = (float*)alloc((size_t)M * 512 * 4);
    float* v    = (float*)alloc((size_t)M * 512 * 4);
    u16*   ao   = (u16*)  alloc((size_t)M * 512 * 2);    // attn out bf16
    float* x2p  = (float*)alloc((size_t)M * 512 * 4);    // pre-LN x2, reused as x3
    float* x2   = (float*)alloc((size_t)M * 512 * 4);
    u16*   x2lb = (u16*)  alloc((size_t)M * 512 * 2);    // LN(x2) bf16 (FFN in)
    u16*   hb   = (u16*)  alloc((size_t)M * 2048 * 2);   // gelu hidden bf16

    auto cvt = [&](const float* src, u16* dst, int n) {
        cvt_bf16_kernel<<<(n + 255) / 256, 256, 0, stream>>>(src, dst, n);
    };
    cvt(x, xb, M * 512);
    cvt(W_in, Winb, 1024 * 512);
    cvt(W_state, Wstb, 512 * 512);
    cvt(Wq, Wqb, 512 * 512);  cvt(Wk, Wkb, 512 * 512);
    cvt(Wv, Wvb, 512 * 512);  cvt(Wp, Wpb, 512 * 512);
    cvt(W1, W1b, 2048 * 512); cvt(W2, W2b, 512 * 2048);

    auto gemm = [&](const u16* A, const u16* W, int N, int K, int mode, float scale,
                    const float* res, const float* bias, float* of, u16* ob) {
        int waves = (M / 32) * (N / 64);
        gemm_bf16_wmma<<<waves / 8, 256, 0, stream>>>(A, W, M, N, K, mode, scale,
                                                      res, bias, of, ob);
    };

    // 1) gv = x @ W_in^T : u = sigmoid(gv[:,:D]) f32 ; g2 = gv[:,D:] bf16
    gemm(xb, Winb, 1024, 512, 1, 1.0f, nullptr, nullptr, ubuf, g2b);
    // 2) cand = g2 @ W_state^T
    gemm(g2b, Wstb, 512, 512, 0, 1.0f, nullptr, nullptr, cand, nullptr);
    // 3) sequential GLRU scan (cand overwritten with y)
    glru_scan_kernel<<<8, 256, 0, stream>>>(ubuf, cand);
    // 4) x1 = LN(y)
    ln_kernel<<<512, 256, 0, stream>>>(cand, glru_g, glru_b, x1, x1b, nullptr, nullptr);
    // 5) q (pre-scaled by d^-0.5), k, v
    gemm(x1b, Wqb, 512, 512, 0, 0.0883883476483184f, nullptr, nullptr, q, nullptr);
    gemm(x1b, Wkb, 512, 512, 0, 1.0f, nullptr, nullptr, k, nullptr);
    gemm(x1b, Wvb, 512, 512, 0, 1.0f, nullptr, nullptr, v, nullptr);
    // 6) windowed attention -> ao (bf16)
    local_attn_kernel<<<2048, 256, 0, stream>>>(q, k, v, ao);
    // 7) x2pre = x1 + ao @ Wp^T
    gemm(ao, Wpb, 512, 512, 2, 1.0f, x1, nullptr, x2p, nullptr);
    // 8) x2 = LN(x2pre); x2lb = bf16(LN(x2, ffn_g, ffn_b))  (fused double LN)
    ln_kernel<<<512, 256, 0, stream>>>(x2p, attn_g, attn_b, x2, x2lb, ffn_g, ffn_b);
    // 9) h = gelu(x2ln @ W1^T + b1) -> bf16
    gemm(x2lb, W1b, 2048, 512, 3, 1.0f, nullptr, b1, nullptr, hb);
    // 10) x3 = x2 + h @ W2^T + b2  (reuse x2p)
    gemm(hb, W2b, 512, 2048, 4, 1.0f, x2, b2, x2p, nullptr);
    // 11) out = LN(x3)
    ln_kernel<<<512, 256, 0, stream>>>(x2p, out_g, out_b, out, nullptr, nullptr, nullptr);
}
